// GatedGCNLayer_79877801771058
// MI455X (gfx1250) — matrix operand order repaired and checked by reading, not compiled, not run on hardware
//
#include <hip/hip_runtime.h>
#include <math.h>

// ---------------------------------------------------------------------------
// GatedGCN layer, N=50000 nodes, E=600000 edges, D=128.
// All linears except B3 hoisted to node space; WMMA f16->f32 GEMMs with
// register-resident pre-packed f16 weights; edge kernels are gather/scatter
// streams (memory-bound, ~2.5-3 GB ~ 130us at 23.3 TB/s HBM).
// ---------------------------------------------------------------------------

typedef __attribute__((ext_vector_type(16))) _Float16 v16h;
typedef __attribute__((ext_vector_type(8)))  float    v8f;

#define D128 128

__device__ __forceinline__ float sigmoidf_(float x) {
    return 1.0f / (1.0f + __expf(-x));
}

// ---------------------------------------------------------------------------
// f32 -> f16 weight pre-pack (run once per launch; weights are tiny).
// ---------------------------------------------------------------------------
__global__ __launch_bounds__(256) void f32_to_f16_kernel(
    const float* __restrict__ src, _Float16* __restrict__ dst, int n)
{
    const int stride = gridDim.x * blockDim.x;
    for (int i = blockIdx.x * blockDim.x + threadIdx.x; i < n; i += stride)
        dst[i] = (_Float16)src[i];
}

// ---------------------------------------------------------------------------
// Y[rows x 128] = concat(X0,X1)[rows x K] @ W16^T[K x 128] + bias
// W16 is f16 row-major [128][K].  K = 128 (X1 unused) or 256 (concat).
// Block: 256 threads = 8 waves; block computes 16 rows x 128 cols.
// Wave w owns column tile w*16 and:
//   - preloads ALL its B fragments into registers (row-tile invariant),
//   - loops K in steps of 32: 2x ds_load_b128 (A frag) + v_wmma_f32_16x16x32_f16.
// X tile staged to LDS as f16 once per block, shared by all 8 waves.
// Full-tile blocks (all but at most one) take unguarded fast paths.
// ---------------------------------------------------------------------------
template <int K>
__global__ __launch_bounds__(256) void gemm_wmma_f16(
    const float* __restrict__ X0, const float* __restrict__ X1,
    const _Float16* __restrict__ W16, const float* __restrict__ bias,
    float* __restrict__ Y, int rows)
{
    constexpr int NK = K / 32;          // k-tiles
    __shared__ _Float16 tA[16 * K];

    const int tid  = threadIdx.x;
    const int row0 = blockIdx.x * 16;
    const bool full = (row0 + 16) <= rows;

    const int wave = tid >> 5;
    const int lane = tid & 31;
    const int lm   = lane & 15;     // M (A) / N (B,C)
    const int lh   = lane >> 4;     // lane-half selector
    const int n0   = wave * 16;     // output column tile

    // Preload B fragments (contiguous k order, same as A: lane half lh covers K j*32+lh*16..+15,
    // column N = lm; B[k][n] = W[n][k] -> 16 contiguous f16 per lane).
    v16h bfrag[NK];
    #pragma unroll
    for (int j = 0; j < NK; ++j) {
        const _Float16* wp = W16 + (size_t)(n0 + lm) * K + (j << 5) + (lh << 4);
        #pragma unroll
        for (int e2 = 0; e2 < 16; ++e2) bfrag[j][e2] = wp[e2];
    }
    const float bv = bias[n0 + lm];

    // Stage X tile (16 x K), converting f32 -> f16.
    constexpr int total = 16 * K;
    if (full) {
        #pragma unroll
        for (int idx0 = 0; idx0 < total; idx0 += 256) {
            int idx = idx0 + tid;
            int m = idx / K;
            int k = idx - m * K;
            float val = (k < 128 || X1 == nullptr)
                            ? X0[(size_t)(row0 + m) * D128 + (k & 127)]
                            : X1[(size_t)(row0 + m) * D128 + (k - 128)];
            tA[m * K + k] = (_Float16)val;
        }
    } else {
        #pragma unroll
        for (int idx0 = 0; idx0 < total; idx0 += 256) {
            int idx = idx0 + tid;
            int m = idx / K;
            int k = idx - m * K;
            int r = row0 + m;
            if (r >= rows) r = rows - 1;         // tail clamp (harmless reread)
            float val = (k < 128 || X1 == nullptr)
                            ? X0[(size_t)r * D128 + (k & 127)]
                            : X1[(size_t)r * D128 + (k - 128)];
            tA[m * K + k] = (_Float16)val;
        }
    }
    __syncthreads();

    v8f acc = {};
    #pragma unroll
    for (int j = 0; j < NK; ++j) {
        // A fragment (contiguous k order, same as B): element e of lane ->
        //   K = j*32 + lh*16 + e, M = lm
        v16h a;
        #pragma unroll
        for (int i = 0; i < 8; ++i) {
            int kk = (j << 5) + (lh << 4) + (i << 1);
            a[2 * i]     = tA[lm * K + kk];
            a[2 * i + 1] = tA[lm * K + kk + 1];
        }
        acc = __builtin_amdgcn_wmma_f32_16x16x32_f16(
                  false, a, false, bfrag[j], (short)0, acc, false, false);
    }

    // C/D layout: VGPR r -> M = r + 8*lh, N = lm.
    float* yp = Y + (size_t)(row0 + lh * 8) * D128 + n0 + lm;
    if (full) {
        #pragma unroll
        for (int r = 0; r < 8; ++r)
            yp[(size_t)r * D128] = acc[r] + bv;
    } else {
        #pragma unroll
        for (int r = 0; r < 8; ++r) {
            int row = row0 + r + lh * 8;
            if (row < rows)
                Y[(size_t)row * D128 + n0 + lm] = acc[r] + bv;
        }
    }
}

// ---------------------------------------------------------------------------
// Edge gate: hat_eta (in place over B3e) += B1h[src] + B2h[dst];
// sum_sigma[dst] += sigmoid(hat_eta)  (atomic segment sum).
// ---------------------------------------------------------------------------
__global__ __launch_bounds__(256) void edge_gate_kernel(
    float* __restrict__ hat_eta,                 // [E,128] in: B3e, out: hat_eta
    const float* __restrict__ B1h, const float* __restrict__ B2h,
    const int* __restrict__ src, const int* __restrict__ dst,
    float* __restrict__ sum_sigma, int E)
{
    const int total4 = E * (D128 / 4);
    const int stride = gridDim.x * blockDim.x;
    for (int i = blockIdx.x * blockDim.x + threadIdx.x; i < total4; i += stride) {
        const int e  = i >> 5;
        const int c4 = (i & 31) * 4;
        const int s  = src[e];
        const int d  = dst[e];
        const size_t eo  = (size_t)e * D128 + c4;
        const size_t so  = (size_t)s * D128 + c4;
        const size_t dmo = (size_t)d * D128 + c4;

        float4 x  = *(const float4*)(hat_eta + eo);
        float4 g1 = *(const float4*)(B1h + so);
        float4 g2 = *(const float4*)(B2h + dmo);
        x.x += g1.x + g2.x;  x.y += g1.y + g2.y;
        x.z += g1.z + g2.z;  x.w += g1.w + g2.w;
        *(float4*)(hat_eta + eo) = x;

        atomicAdd(&sum_sigma[dmo + 0], sigmoidf_(x.x));
        atomicAdd(&sum_sigma[dmo + 1], sigmoidf_(x.y));
        atomicAdd(&sum_sigma[dmo + 2], sigmoidf_(x.z));
        atomicAdd(&sum_sigma[dmo + 3], sigmoidf_(x.w));
    }
}

// ---------------------------------------------------------------------------
// Edge aggregate: eta = sigmoid(hat_eta) / (sum_sigma[dst] + eps);
// hagg[dst] += eta * v[src];  pagg[dst] += eta * c2p[src].
// hagg/pagg were pre-initialized by the A1/C1 GEMMs.
// ---------------------------------------------------------------------------
__global__ __launch_bounds__(256) void edge_aggregate_kernel(
    const float* __restrict__ hat_eta, const float* __restrict__ sum_sigma,
    const float* __restrict__ vbuf, const float* __restrict__ c2p,
    const int* __restrict__ src, const int* __restrict__ dst,
    float* __restrict__ hagg, float* __restrict__ pagg, int E)
{
    const int total4 = E * (D128 / 4);
    const int stride = gridDim.x * blockDim.x;
    for (int i = blockIdx.x * blockDim.x + threadIdx.x; i < total4; i += stride) {
        const int e  = i >> 5;
        const int c4 = (i & 31) * 4;
        const int s  = src[e];
        const int d  = dst[e];
        const size_t eo  = (size_t)e * D128 + c4;
        const size_t so  = (size_t)s * D128 + c4;
        const size_t dmo = (size_t)d * D128 + c4;

        float4 he = *(const float4*)(hat_eta + eo);
        float4 ss = *(const float4*)(sum_sigma + dmo);
        float4 vv = *(const float4*)(vbuf + so);
        float4 cc = *(const float4*)(c2p + so);

        float eta0 = sigmoidf_(he.x) / (ss.x + 1e-6f);
        float eta1 = sigmoidf_(he.y) / (ss.y + 1e-6f);
        float eta2 = sigmoidf_(he.z) / (ss.z + 1e-6f);
        float eta3 = sigmoidf_(he.w) / (ss.w + 1e-6f);

        atomicAdd(&hagg[dmo + 0], eta0 * vv.x);
        atomicAdd(&hagg[dmo + 1], eta1 * vv.y);
        atomicAdd(&hagg[dmo + 2], eta2 * vv.z);
        atomicAdd(&hagg[dmo + 3], eta3 * vv.w);

        atomicAdd(&pagg[dmo + 0], eta0 * cc.x);
        atomicAdd(&pagg[dmo + 1], eta1 * cc.y);
        atomicAdd(&pagg[dmo + 2], eta2 * cc.z);
        atomicAdd(&pagg[dmo + 3], eta3 * cc.w);
    }
}

// ---------------------------------------------------------------------------
// BatchNorm stats: per-column sum and sum-of-squares over `rows`.
// ---------------------------------------------------------------------------
__global__ __launch_bounds__(256) void bn_stats_kernel(
    const float* __restrict__ X, int rows,
    float* __restrict__ gsum, float* __restrict__ gsumsq)
{
    __shared__ float s1[256];
    __shared__ float s2[256];
    const int col  = threadIdx.x & 127;
    const int half = threadIdx.x >> 7;
    float a = 0.f, b = 0.f;
    for (int r = blockIdx.x * 2 + half; r < rows; r += gridDim.x * 2) {
        float x = X[(size_t)r * D128 + col];
        a += x;
        b += x * x;
    }
    s1[threadIdx.x] = a;
    s2[threadIdx.x] = b;
    __syncthreads();
    if (half == 0) {
        atomicAdd(&gsum[col],   s1[threadIdx.x] + s1[threadIdx.x + 128]);
        atomicAdd(&gsumsq[col], s2[threadIdx.x] + s2[threadIdx.x + 128]);
    }
}

// out = residual + relu(bn(x))  (in place over x)
__global__ __launch_bounds__(256) void finalize_bn_relu_kernel(
    const float* __restrict__ resid, float* __restrict__ x_inout,
    const float* __restrict__ gamma, const float* __restrict__ beta,
    const float* __restrict__ gsum, const float* __restrict__ gsumsq, int rows)
{
    const int total = rows * D128;
    const int stride = gridDim.x * blockDim.x;
    const float inv_n = 1.0f / (float)rows;
    for (int i = blockIdx.x * blockDim.x + threadIdx.x; i < total; i += stride) {
        const int c = i & 127;
        float m   = gsum[c] * inv_n;
        float var = gsumsq[c] * inv_n - m * m;
        float y   = gamma[c] * (x_inout[i] - m) * rsqrtf(var + 1e-5f) + beta[c];
        x_inout[i] = resid[i] + fmaxf(y, 0.0f);
    }
}

// out = residual + tanh(x)  (in place over x)
__global__ __launch_bounds__(256) void finalize_tanh_kernel(
    const float* __restrict__ resid, float* __restrict__ x_inout, int rows)
{
    const int total = rows * D128;
    const int stride = gridDim.x * blockDim.x;
    for (int i = blockIdx.x * blockDim.x + threadIdx.x; i < total; i += stride) {
        x_inout[i] = resid[i] + tanhf(x_inout[i]);
    }
}

// ---------------------------------------------------------------------------
extern "C" void kernel_launch(void* const* d_in, const int* in_sizes, int n_in,
                              void* d_out, int out_size, void* d_ws, size_t ws_size,
                              hipStream_t stream)
{
    const float* h   = (const float*)d_in[0];
    const float* e   = (const float*)d_in[1];
    const float* p   = (const float*)d_in[2];
    const int*   src = (const int*)  d_in[3];
    const int*   dst = (const int*)  d_in[4];
    const float* A1w = (const float*)d_in[5];
    const float* A1b = (const float*)d_in[6];
    const float* A2w = (const float*)d_in[7];
    const float* A2b = (const float*)d_in[8];
    const float* B1w = (const float*)d_in[9];
    const float* B1b = (const float*)d_in[10];
    const float* B2w = (const float*)d_in[11];
    const float* B2b = (const float*)d_in[12];
    const float* B3w = (const float*)d_in[13];
    const float* B3b = (const float*)d_in[14];
    const float* C1w = (const float*)d_in[15];
    const float* C1b = (const float*)d_in[16];
    const float* C2w = (const float*)d_in[17];
    const float* C2b = (const float*)d_in[18];
    const float* bn_h_g = (const float*)d_in[19];
    const float* bn_h_b = (const float*)d_in[20];
    const float* bn_e_g = (const float*)d_in[21];
    const float* bn_e_b = (const float*)d_in[22];

    const int N = in_sizes[0] / D128;
    const int E = in_sizes[1] / D128;

    // d_out layout: [h_out N*128 | e_out E*128 | p_out N*128]
    float* out     = (float*)d_out;
    float* hagg    = out;                          // A1h + scatter, then h_out
    float* hat_eta = out + (size_t)N * D128;       // B3e -> hat_eta, then e_out
    float* pagg    = hat_eta + (size_t)E * D128;   // C1p + scatter, then p_out

    // Workspace (~128 MB f32 + ~260 KB f16 weights)
    float* ws       = (float*)d_ws;
    float* B1h      = ws;  ws += (size_t)N * D128;
    float* B2h      = ws;  ws += (size_t)N * D128;
    float* vbuf     = ws;  ws += (size_t)N * D128;   // A2([h;p])
    float* c2p      = ws;  ws += (size_t)N * D128;   // C2 p
    float* ssum     = ws;  ws += (size_t)N * D128;   // segment sum of sigma
    float* st_h_sum = ws;  ws += D128;
    float* st_h_sq  = ws;  ws += D128;
    float* st_e_sum = ws;  ws += D128;
    float* st_e_sq  = ws;  ws += D128;

    _Float16* w16 = (_Float16*)ws;
    _Float16* A1w16 = w16;  w16 += 128 * 128;
    _Float16* A2w16 = w16;  w16 += 128 * 256;
    _Float16* B1w16 = w16;  w16 += 128 * 128;
    _Float16* B2w16 = w16;  w16 += 128 * 128;
    _Float16* B3w16 = w16;  w16 += 128 * 128;
    _Float16* C1w16 = w16;  w16 += 128 * 128;
    _Float16* C2w16 = w16;  w16 += 128 * 128;

    const dim3 blk(256);

    // Pre-pack weights to f16 (tiny, once per launch).
    f32_to_f16_kernel<<<64, blk, 0, stream>>>(A1w, A1w16, 128 * 128);
    f32_to_f16_kernel<<<128, blk, 0, stream>>>(A2w, A2w16, 128 * 256);
    f32_to_f16_kernel<<<64, blk, 0, stream>>>(B1w, B1w16, 128 * 128);
    f32_to_f16_kernel<<<64, blk, 0, stream>>>(B2w, B2w16, 128 * 128);
    f32_to_f16_kernel<<<64, blk, 0, stream>>>(B3w, B3w16, 128 * 128);
    f32_to_f16_kernel<<<64, blk, 0, stream>>>(C1w, C1w16, 128 * 128);
    f32_to_f16_kernel<<<64, blk, 0, stream>>>(C2w, C2w16, 128 * 128);

    hipMemsetAsync(ssum, 0, (size_t)N * D128 * sizeof(float), stream);
    hipMemsetAsync(st_h_sum, 0, 4 * D128 * sizeof(float), stream);

    const int nG = (N + 15) / 16;
    const int eG = (E + 15) / 16;

    // Node-space GEMMs (WMMA f16 -> f32)
    gemm_wmma_f16<128><<<nG, blk, 0, stream>>>(h, nullptr, B1w16, B1b, B1h, N);
    gemm_wmma_f16<128><<<nG, blk, 0, stream>>>(h, nullptr, B2w16, B2b, B2h, N);
    gemm_wmma_f16<256><<<nG, blk, 0, stream>>>(h, p,       A2w16, A2b, vbuf, N);
    gemm_wmma_f16<128><<<nG, blk, 0, stream>>>(p, nullptr, C2w16, C2b, c2p, N);
    gemm_wmma_f16<128><<<nG, blk, 0, stream>>>(h, nullptr, A1w16, A1b, hagg, N);
    gemm_wmma_f16<128><<<nG, blk, 0, stream>>>(p, nullptr, C1w16, C1b, pagg, N);
    // Edge-space GEMM: B3 e  (written straight into e_out slice of d_out)
    gemm_wmma_f16<128><<<eG, blk, 0, stream>>>(e, nullptr, B3w16, B3b, hat_eta, E);

    // Edge streams
    edge_gate_kernel<<<4096, blk, 0, stream>>>(hat_eta, B1h, B2h, src, dst, ssum, E);
    edge_aggregate_kernel<<<4096, blk, 0, stream>>>(hat_eta, ssum, vbuf, c2p,
                                                    src, dst, hagg, pagg, E);

    // BatchNorm stats + fused finalize (BN + ReLU/tanh + residual), in place
    bn_stats_kernel<<<1024, blk, 0, stream>>>(hagg, N, st_h_sum, st_h_sq);
    bn_stats_kernel<<<2048, blk, 0, stream>>>(hat_eta, E, st_e_sum, st_e_sq);

    finalize_bn_relu_kernel<<<2048, blk, 0, stream>>>(h, hagg, bn_h_g, bn_h_b,
                                                      st_h_sum, st_h_sq, N);
    finalize_bn_relu_kernel<<<4096, blk, 0, stream>>>(e, hat_eta, bn_e_g, bn_e_b,
                                                      st_e_sum, st_e_sq, E);
    finalize_tanh_kernel<<<2048, blk, 0, stream>>>(p, pagg, N);
}
